// genpatchwithMask_80788334837909
// MI455X (gfx1250) — matrix operand order repaired and checked by reading, not compiled, not run on hardware
//
#include <hip/hip_runtime.h>
#include <hip/hip_bf16.h>

typedef _Float16 half16 __attribute__((ext_vector_type(16)));
typedef float    float8 __attribute__((ext_vector_type(8)));

#define HH   256          // image H = W
#define PP   225          // pooled size = 256 - 31
#define PN   (PP * PP)    // 50625
#define NB   8
#define NMAP 32           // B * 2 classes
#define KSEL 2

// output segment offsets (flat f32, reference return order)
#define N_CLS   65536      // [2,2,8,2,32,32]
#define OFF_FEAT 65536
#define N_FEAT  2097152    // [2,2,8,64,32,32]
#define OFF_VALS 2162688   // [2,2,8]
#define OFF_PES  2162720   // [2,2,8,1,32,32]
#define OFF_TRUE 2195488
#define OFF_OXY  2228256   // [2,2,8,2,2]
#define N_TOTAL  2228384

// ---- workspace layout (bytes) ----
#define WS_COLSUM 0                         // 32 * 225 * 256 f32
#define WS_POOLED (NMAP * PP * HH * 4)      // 32 * 225 * 225 f32
#define WS_SEL    (WS_POOLED + NMAP * PP * PP * 4)

// softmax over the 2-channel axis; 1-ulp v_rcp_f32 instead of IEEE divide
// (feeds an argmax + avg-pool; 1e-7 relative error is irrelevant here).
__device__ __forceinline__ float softmax_prob(const float* __restrict__ infeat,
                                              int bb, int c, int j, int x) {
  size_t base = ((size_t)bb * 2) * HH * HH + (size_t)j * HH + x;
  float v0 = infeat[base];
  float v1 = infeat[base + (size_t)HH * HH];
  float self  = (c == 0) ? v0 : v1;
  float other = (c == 0) ? v1 : v0;
  return __builtin_amdgcn_rcpf(1.0f + __expf(other - self));
}

// A-matrix (16x32 f16) K index for vgpr-half v (0..7), lane-half h (0/1)
__device__ __forceinline__ int ksA(int v, int h) {
  return (v < 4) ? (h * 8 + 2 * v) : (16 + h * 8 + 2 * (v - 4));
}
// B-matrix (32x16 f16) K index
__device__ __forceinline__ int ksB(int v, int h) { return (v < 4) ? (h * 8 + 2 * v) : (16 + h * 8 + 2 * (v - 4)); }

// ---------------------------------------------------------------------------
// Kernel 1: fused softmax + vertical 32-wide box sum via banded WMMA.
// colsum[map][y][x] = sum_{j=y}^{y+31} softmax(infeat)[map][j][x], y<225.
// One wave per 16x16 output tile. grid = 32 maps * 15 ytiles * 16 xtiles.
// NOTE: for every executed K-chunk, kbase <= 224 and lane K offset <= 30,
// so all j indices are < 256 — no per-lane guards needed (keeps EXEC all-1s
// with no divergent branches around the WMMAs).
// ---------------------------------------------------------------------------
__global__ void __launch_bounds__(32)
vpool_kernel(const float* __restrict__ infeat, float* __restrict__ colsum) {
  int wid = blockIdx.x;
  int xt = wid & 15; wid >>= 4;
  int yt = wid % 15; int map = wid / 15;
  int bb = map >> 1, c = map & 1;
  int lane = threadIdx.x;
  int h = lane >> 4, lan = lane & 15;
  int y0 = yt * 16, x0 = xt * 16;
  int yA = y0 + lan;   // A-matrix row (M)
  int xB = x0 + lan;   // B-matrix col (N)

  float8 acc = {};
  int klo = y0 & ~31;
  for (int cidx = 0; cidx < 2; ++cidx) {
    int kbase = klo + cidx * 32;
    if (kbase >= HH) break;             // uniform per wave
    half16 A, Bhi, Blo;
#pragma unroll
    for (int v = 0; v < 8; ++v) {
      int ka = kbase + ksA(v, h);
      A[2 * v]     = (_Float16)((yA < PP && ka     >= yA && ka     < yA + 32) ? 1.0f : 0.0f);
      A[2 * v + 1] = (_Float16)((yA < PP && ka + 1 >= yA && ka + 1 < yA + 32) ? 1.0f : 0.0f);
      int kb = kbase + ksB(v, h);
      float pa = softmax_prob(infeat, bb, c, kb,     xB);
      float pb = softmax_prob(infeat, bb, c, kb + 1, xB);
      _Float16 ha = (_Float16)pa, hb = (_Float16)pb;
      Bhi[2 * v] = ha;                         Bhi[2 * v + 1] = hb;
      Blo[2 * v] = (_Float16)(pa - (float)ha); Blo[2 * v + 1] = (_Float16)(pb - (float)hb);
    }
    // double-f16: acc += Band * (hi + lo)  -> ~f32-accurate box sum
    acc = __builtin_amdgcn_wmma_f32_16x16x32_f16(false, A, false, Bhi, (short)0, acc, false, false);
    acc = __builtin_amdgcn_wmma_f32_16x16x32_f16(false, A, false, Blo, (short)0, acc, false, false);
  }
#pragma unroll
  for (int r = 0; r < 8; ++r) {
    int yy = y0 + h * 8 + r;
    if (yy < PP) colsum[((size_t)map * PP + yy) * HH + x0 + lan] = acc[r];
  }
}

// ---------------------------------------------------------------------------
// Kernel 2: horizontal 32-wide box sum via banded WMMA, /1024 at store.
// pooled[map][y][x] = (1/1024) * sum_{j=x}^{x+31} colsum[map][y][j].
// grid = 32 maps * 15 ytiles * 15 xtiles. Same index-range argument as above:
// all K indices < 256, only the yA<PP row clamp remains (select, not branch).
// ---------------------------------------------------------------------------
__global__ void __launch_bounds__(32)
hpool_kernel(const float* __restrict__ colsum, float* __restrict__ pooled) {
  int wid = blockIdx.x;
  int xt = wid % 15; wid /= 15;
  int yt = wid % 15; int map = wid / 15;
  int lane = threadIdx.x;
  int h = lane >> 4, lan = lane & 15;
  int y0 = yt * 16, xo0 = xt * 16;
  int yA = y0 + lan;    // A row (M) = output y
  int xN = xo0 + lan;   // B col (N) = output x

  // clamp the row pointer (rows >= 225 contribute via zeroed values)
  float rowScale = (yA < PP) ? 1.0f : 0.0f;
  const float* row = colsum + ((size_t)map * PP + (yA < PP ? yA : 0)) * HH;

  float8 acc = {};
  int klo = xo0 & ~31;
  for (int cidx = 0; cidx < 2; ++cidx) {
    int kbase = klo + cidx * 32;
    if (kbase >= HH) break;
    half16 Ahi, Alo, Bm;
#pragma unroll
    for (int v = 0; v < 8; ++v) {
      int ka = kbase + ksA(v, h);
      float va = row[ka]     * rowScale;
      float vb = row[ka + 1] * rowScale;
      _Float16 ha = (_Float16)va, hb = (_Float16)vb;
      Ahi[2 * v] = ha;                          Ahi[2 * v + 1] = hb;
      Alo[2 * v] = (_Float16)(va - (float)ha);  Alo[2 * v + 1] = (_Float16)(vb - (float)hb);
      int kb = kbase + ksB(v, h);
      Bm[2 * v]     = (_Float16)((kb     >= xN && kb     < xN + 32) ? 1.0f : 0.0f);
      Bm[2 * v + 1] = (_Float16)((kb + 1 >= xN && kb + 1 < xN + 32) ? 1.0f : 0.0f);
    }
    acc = __builtin_amdgcn_wmma_f32_16x16x32_f16(false, Ahi, false, Bm, (short)0, acc, false, false);
    acc = __builtin_amdgcn_wmma_f32_16x16x32_f16(false, Alo, false, Bm, (short)0, acc, false, false);
  }
  const float inv = 1.0f / 1024.0f;
#pragma unroll
  for (int r = 0; r < 8; ++r) {
    int yy = y0 + h * 8 + r;
    int xx = xo0 + lan;
    if (yy < PP && xx < PP) pooled[((size_t)map * PP + yy) * PP + xx] = acc[r] * inv;
  }
}

// ---------------------------------------------------------------------------
// Kernel 3: greedy k=2 argmax with 32x32 window suppression per map.
// One 256-thread block per map; writes vals + oxy outputs and (px,py) table.
// ---------------------------------------------------------------------------
__global__ void __launch_bounds__(256)
select_kernel(const float* __restrict__ pooled, int* __restrict__ sel,
              float* __restrict__ out) {
  int map = blockIdx.x;
  int bb = map >> 1, c = map & 1;
  const float* pm = pooled + (size_t)map * PN;
  __shared__ float sv[256];
  __shared__ int   si[256];
  __shared__ int   swin;
  int tid = threadIdx.x;
  int spx[KSEL], spy[KSEL];
  int nsup = 0;

  for (int kk = 0; kk < KSEL; ++kk) {
    float best = -1.0f;   // pooled softmax avgs are > 0; masked -> 0
    int   bidx = 0;
    for (int idx = tid; idx < PN; idx += 256) {
      int yy = idx / PP, xx = idx % PP;
      float v = pm[idx];
      for (int s = 0; s < nsup; ++s) {
        bool in = (yy >= spy[s] - 16) && (yy < spy[s] + 16) &&
                  (xx >= spx[s] - 16) && (xx < spx[s] + 16);
        if (in) v = 0.0f;                 // mirrors pm * mask
      }
      if (v > best) { best = v; bidx = idx; }   // keeps lowest idx on ties
    }
    sv[tid] = best; si[tid] = bidx;
    __syncthreads();
    for (int off = 128; off > 0; off >>= 1) {
      if (tid < off) {
        float v2 = sv[tid + off]; int i2 = si[tid + off];
        if (v2 > sv[tid] || (v2 == sv[tid] && i2 < si[tid])) { sv[tid] = v2; si[tid] = i2; }
      }
      __syncthreads();
    }
    if (tid == 0) {
      swin = si[0];
      int py = swin / PP, px = swin % PP;
      sel[(map * KSEL + kk) * 2 + 0] = px;
      sel[(map * KSEL + kk) * 2 + 1] = py;
      int vi = (c * KSEL + kk) * NB + bb;       // [2,k,B]
      out[OFF_VALS + vi] = sv[0];
      size_t ob = (size_t)OFF_OXY + (size_t)vi * 4; // [2,k,B,2,2]
      out[ob + 0] = (float)px; out[ob + 1] = (float)(px + 31);
      out[ob + 2] = (float)py; out[ob + 3] = (float)(py + 31);
    }
    __syncthreads();
    int w = swin;
    spy[nsup] = w / PP; spx[nsup] = w % PP; ++nsup;
    __syncthreads();
  }
}

// ---------------------------------------------------------------------------
// Kernel 4: patch extraction (gather) into the concatenated output.
// ---------------------------------------------------------------------------
__global__ void __launch_bounds__(256)
extract_kernel(const float* __restrict__ infeat, const float* __restrict__ labelTpesudo,
               const float* __restrict__ labelT, const float* __restrict__ FeatureDA,
               const int* __restrict__ sel, float* __restrict__ out) {
  int gid = blockIdx.x * 256 + threadIdx.x;
  if (gid >= N_TOTAL) return;

  int cls, kk, bb, ch, yy, xx;
  const float* src;
  size_t srcoff;

  if (gid < OFF_FEAT) {                       // classifer_p [2,2,8,2,32,32]
    int t = gid;
    xx = t & 31; t >>= 5; yy = t & 31; t >>= 5;
    ch = t & 1;  t >>= 1; bb = t & 7;  t >>= 3;
    kk = t & 1;  t >>= 1; cls = t & 1;
    int px = sel[((bb * 2 + cls) * KSEL + kk) * 2 + 0];
    int py = sel[((bb * 2 + cls) * KSEL + kk) * 2 + 1];
    src = infeat;
    srcoff = (((size_t)bb * 2 + ch) * HH + (py + yy)) * HH + (px + xx);
  } else if (gid < OFF_VALS) {                // feat_p [2,2,8,64,32,32]
    int t = gid - OFF_FEAT;
    xx = t & 31; t >>= 5; yy = t & 31; t >>= 5;
    ch = t & 63; t >>= 6; bb = t & 7;  t >>= 3;
    kk = t & 1;  t >>= 1; cls = t & 1;
    int px = sel[((bb * 2 + cls) * KSEL + kk) * 2 + 0];
    int py = sel[((bb * 2 + cls) * KSEL + kk) * 2 + 1];
    src = FeatureDA;
    srcoff = (((size_t)bb * 64 + ch) * HH + (py + yy)) * HH + (px + xx);
  } else if (gid < OFF_PES) {
    return;                                   // vals written by select_kernel
  } else if (gid < OFF_TRUE) {                // pesudo_p [2,2,8,1,32,32]
    int t = gid - OFF_PES;
    xx = t & 31; t >>= 5; yy = t & 31; t >>= 5;
    bb = t & 7;  t >>= 3; kk = t & 1;  t >>= 1; cls = t & 1;
    int px = sel[((bb * 2 + cls) * KSEL + kk) * 2 + 0];
    int py = sel[((bb * 2 + cls) * KSEL + kk) * 2 + 1];
    src = labelTpesudo;
    srcoff = ((size_t)bb * HH + (py + yy)) * HH + (px + xx);
  } else if (gid < OFF_OXY) {                 // true_p [2,2,8,1,32,32]
    int t = gid - OFF_TRUE;
    xx = t & 31; t >>= 5; yy = t & 31; t >>= 5;
    bb = t & 7;  t >>= 3; kk = t & 1;  t >>= 1; cls = t & 1;
    int px = sel[((bb * 2 + cls) * KSEL + kk) * 2 + 0];
    int py = sel[((bb * 2 + cls) * KSEL + kk) * 2 + 1];
    src = labelT;
    srcoff = ((size_t)bb * HH + (py + yy)) * HH + (px + xx);
  } else {
    return;                                   // oxy written by select_kernel
  }
  out[gid] = src[srcoff];
}

extern "C" void kernel_launch(void* const* d_in, const int* in_sizes, int n_in,
                              void* d_out, int out_size, void* d_ws, size_t ws_size,
                              hipStream_t stream) {
  const float* infeat        = (const float*)d_in[0];
  const float* labelTpesudo  = (const float*)d_in[1];
  const float* labelT        = (const float*)d_in[2];
  const float* FeatureDA     = (const float*)d_in[3];
  // d_in[4] holds k; setup fixes k=2 and out_size depends on it -> hardcoded.

  float* out = (float*)d_out;
  char*  ws  = (char*)d_ws;
  float* colsum = (float*)(ws + WS_COLSUM);
  float* pooled = (float*)(ws + WS_POOLED);
  int*   sel    = (int*)(ws + WS_SEL);

  vpool_kernel<<<NMAP * 15 * 16, 32, 0, stream>>>(infeat, colsum);
  hpool_kernel<<<NMAP * 15 * 15, 32, 0, stream>>>(colsum, pooled);
  select_kernel<<<NMAP, 256, 0, stream>>>(pooled, sel, out);
  extract_kernel<<<(N_TOTAL + 255) / 256, 256, 0, stream>>>(
      infeat, labelTpesudo, labelT, FeatureDA, sel, out);
}
